// GraphAttentionLayer_4750233829805
// MI455X (gfx1250) — matrix operand order repaired and checked by reading, not compiled, not run on hardware
//
#include <hip/hip_runtime.h>
#include <hip/hip_bf16.h>

#define N      8192
#define FIN    512
#define FOUT   64
#define BETA   0.5f
#define ALPHA  0.2f
#define NCH    (N / 32)   // bitmask words per row (ballot-order layout)

typedef __attribute__((ext_vector_type(16))) __bf16 v16bf;
typedef __attribute__((ext_vector_type(8)))  float  v8f;
typedef __attribute__((ext_vector_type(8)))  unsigned short vus8;

union FragU { vus8 h[2]; v16bf v; };

__device__ __forceinline__ unsigned short f2bf(float x) {
  unsigned int u = __float_as_uint(x);
  u += 0x7FFFu + ((u >> 16) & 1u);          // round-to-nearest-even
  return (unsigned short)(u >> 16);
}

__device__ __forceinline__ unsigned int ballot32(bool p) {
#if __has_builtin(__builtin_amdgcn_ballot_w32)
  return __builtin_amdgcn_ballot_w32(p);
#else
  return (unsigned int)__ballot(p);
#endif
}

// Bitmask layout (ballot order): word (r, 4c+e) bit l  <=>  adj[r][128c + 4l + e]
// For key j: word = (j>>7)*4 + (j&3), bit = (j>>2)&31.

// ---------------- Kernel A: Wh = h @ W  (fp32 + transposed bf16 copy) ----------------
__global__ void k_gemm_wh(const float* __restrict__ h, const float* __restrict__ W,
                          float* __restrict__ Wh, unsigned short* __restrict__ WhbfT) {
  int idx = blockIdx.x * blockDim.x + threadIdx.x;   // N*FOUT threads
  int row = idx >> 6, col = idx & 63;
  const float* hr = h + (size_t)row * FIN;           // broadcast across 64 threads
  float acc = 0.f;
#pragma unroll 4
  for (int k = 0; k < FIN; ++k)
    acc = fmaf(hr[k], W[k * FOUT + col], acc);       // W row reads coalesced
  Wh[idx] = acc;
  WhbfT[(size_t)col * N + row] = f2bf(acc);          // [FOUT][N] for contiguous K-runs
}

// ---------------- Kernel A2: s_i = Wh_i . a1 ; u_i = Wh_i . a2 ----------------
__global__ void k_su(const float* __restrict__ Wh, const float* __restrict__ a,
                     float* __restrict__ s, float* __restrict__ u) {
  int r    = (blockIdx.x * blockDim.x + threadIdx.x) >> 5;   // one wave per row
  int lane = threadIdx.x & 31;
  const float* wr = Wh + (size_t)r * FOUT;
  float w0 = wr[lane], w1 = wr[lane + 32];
  float ss = w0 * a[lane]        + w1 * a[lane + 32];
  float uu = w0 * a[FOUT + lane] + w1 * a[FOUT + lane + 32];
#pragma unroll
  for (int off = 16; off; off >>= 1) {
    ss += __shfl_xor(ss, off, 32);
    uu += __shfl_xor(uu, off, 32);
  }
  if (lane == 0) { s[r] = ss; u[r] = uu; }
}

// ---------------- Kernel B: t = (0.5*M + 0.5*adj) @ u, adj -> ballot-order bitmask ----
// The HBM-bound pass (512 MB). b128 loads + u[] served from LDS so the vmem pipe
// carries only HBM traffic (2 x b128 per lane-iteration).
__global__ void __launch_bounds__(256)
k_t_bits(const int* __restrict__ adj, const float* __restrict__ M,
         const float* __restrict__ u,
         float* __restrict__ t, uint4* __restrict__ bits4) {
  __shared__ __align__(16) float uS[N];              // 32 KB: whole u vector
  {
    const float4* u4 = (const float4*)u;
    float4* uS4 = (float4*)uS;
#pragma unroll
    for (int q = 0; q < 8; ++q)
      uS4[threadIdx.x + 256 * q] = u4[threadIdx.x + 256 * q];
  }
  __syncthreads();

  int r    = (blockIdx.x * blockDim.x + threadIdx.x) >> 5;   // one wave per row
  int lane = threadIdx.x & 31;
  const int4*   a4 = (const int4*)(adj + (size_t)r * N);
  const float4* m4 = (const float4*)(M + (size_t)r * N);
  const float4* uS4 = (const float4*)uS;
  float acc = 0.f;
  for (int c = 0; c < N / 128; ++c) {
    int idx = c * 32 + lane;
    if (c + 8 < N / 128) {                              // prefetch 4KB ahead
      __builtin_prefetch(m4 + idx + 256, 0, 0);
      __builtin_prefetch(a4 + idx + 256, 0, 0);
    }
    int4   av = a4[idx];
    float4 mv = m4[idx];
    float4 uv = uS4[idx];                               // ds_load_b128 (LDS pipe)
    acc = fmaf(BETA * mv.x + (1.f - BETA) * (float)av.x, uv.x, acc);
    acc = fmaf(BETA * mv.y + (1.f - BETA) * (float)av.y, uv.y, acc);
    acc = fmaf(BETA * mv.z + (1.f - BETA) * (float)av.z, uv.z, acc);
    acc = fmaf(BETA * mv.w + (1.f - BETA) * (float)av.w, uv.w, acc);
    unsigned int b0 = ballot32(av.x != 0);
    unsigned int b1 = ballot32(av.y != 0);
    unsigned int b2 = ballot32(av.z != 0);
    unsigned int b3 = ballot32(av.w != 0);
    if (lane == 0) bits4[(size_t)r * (N / 128) + c] = make_uint4(b0, b1, b2, b3);
  }
#pragma unroll
  for (int off = 16; off; off >>= 1) acc += __shfl_xor(acc, off, 32);
  if (lane == 0) t[r] = acc;
}

// ---------------- Kernel C: per-row masked online softmax stats (m_i, 1/Z_i) ----------
__global__ void k_rowstats(const uint4* __restrict__ bits4,
                           const float* __restrict__ s, const float* __restrict__ t,
                           float* __restrict__ m, float* __restrict__ invZ) {
  int r    = (blockIdx.x * blockDim.x + threadIdx.x) >> 5;   // one wave per row
  int lane = threadIdx.x & 31;
  float sr = s[r];
  float mx = -3.0e38f, sm = 0.f;
  const uint4*  br = bits4 + (size_t)r * (N / 128);
  const float4* t4 = (const float4*)t;
  for (int c = 0; c < N / 128; ++c) {
    uint4  w  = br[c];
    float4 tv = t4[c * 32 + lane];      // keys 128c + 4*lane + {0..3}
#pragma unroll
    for (int e = 0; e < 4; ++e) {
      unsigned int we = (e == 0) ? w.x : (e == 1) ? w.y : (e == 2) ? w.z : w.w;
      if ((we >> lane) & 1u) {
        float tj = (e == 0) ? tv.x : (e == 1) ? tv.y : (e == 2) ? tv.z : tv.w;
        float ev = sr + tj;
        ev = fmaxf(ev, ALPHA * ev);                  // leaky relu
        if (ev > mx) { sm *= __expf(mx - ev); mx = ev; }
        sm += __expf(ev - mx);
      }
    }
  }
#pragma unroll
  for (int off = 16; off; off >>= 1) {               // merge lane-local (max,sum)
    float mo = __shfl_xor(mx, off, 32);
    float so = __shfl_xor(sm, off, 32);
    float M2 = fmaxf(mx, mo);
    sm = sm * __expf(mx - M2) + so * __expf(mo - M2);
    mx = M2;
  }
  if (lane == 0) { m[r] = mx; invZ[r] = sm > 0.f ? 1.f / sm : 0.f; }
}

// ---------------- Kernel D: h' = elu( softmax(att) @ Wh ) via bf16 WMMA ----------------
// 16 output rows/block; 4 waves own 16-col tiles; K steps of 64 keys -> 2 WMMAs/wave/step.
// Bitmask rows (16 KB) and t (32 KB) staged in LDS once; probability build is
// branch-free and overflow-safe: p = bit * exp(min(lrelu(s+t) - m, 0)).
__global__ void __launch_bounds__(128)
k_aggregate(const unsigned int* __restrict__ bits, const float* __restrict__ s,
            const float* __restrict__ t, const float* __restrict__ m,
            const float* __restrict__ invZ,
            const unsigned short* __restrict__ WhbfT,
            float* __restrict__ out) {
  __shared__ __align__(16) unsigned int   sBits[16 * NCH];   // 16 KB
  __shared__ __align__(16) float          sT[N];             // 32 KB
  __shared__ __align__(16) unsigned short pA[16 * 64];       // 2 KB prob tile [row][K]
  __shared__ __align__(16) unsigned short pB[64 * 64];       // 8 KB Wh tile  [n][K]
  __shared__ float sS[16], sM[16], sZ[16];

  int tid = threadIdx.x;
  int i0  = blockIdx.x * 16;
  if (tid < 16) { sS[tid] = s[i0 + tid]; sM[tid] = m[i0 + tid]; sZ[tid] = invZ[i0 + tid]; }
  {                                      // stage 16 bitmask rows (contiguous 16 KB)
    const uint4* src4 = (const uint4*)(bits + (size_t)i0 * NCH);
    uint4* dst4 = (uint4*)sBits;
    uint4 tb[8];
#pragma unroll
    for (int q = 0; q < 8; ++q) tb[q] = src4[tid + 128 * q];
#pragma unroll
    for (int q = 0; q < 8; ++q) dst4[tid + 128 * q] = tb[q];
  }
  {                                      // stage full t vector (32 KB)
    const float4* src4 = (const float4*)t;
    float4* dst4 = (float4*)sT;
#pragma unroll
    for (int q = 0; q < 16; ++q) dst4[tid + 128 * q] = src4[tid + 128 * q];
  }
  __syncthreads();

  int lane = tid & 31;
  int wv   = tid >> 5;        // wave id -> N-tile [16*wv, 16*wv+16)
  int rr   = lane & 15;
  int half = lane >> 4;       // K-half select per ISA fragment layout
  int kk   = tid & 63;        // this thread's key offset within the 64-chunk

  float rs[8], rm[8];         // row constants for this thread's 8 rows
#pragma unroll
  for (int q = 0; q < 8; ++q) {
    int row = (tid >> 6) + 2 * q;
    rs[q] = sS[row]; rm[q] = sM[row];
  }

  const uint4* Wt4 = (const uint4*)WhbfT;
  uint4* pB4 = (uint4*)pB;

  v8f acc = {};
  for (int k0 = 0; k0 < N; k0 += 64) {
    __syncthreads();          // protect LDS against previous iteration's readers

    // --- probability tile (branch-free, all operands LDS/registers) ---
    int j  = k0 + kk;
    int cw = ((j >> 7) << 2) + (kk & 3);      // ballot-order word index
    int bp = (j >> 2) & 31;                   // bit position
    float tj = sT[j];
#pragma unroll
    for (int q = 0; q < 8; ++q) {
      int row = (tid >> 6) + 2 * q;
      unsigned int w = sBits[row * NCH + cw];
      float bitf = (float)((w >> bp) & 1u);
      float ev = rs[q] + tj;
      ev = fmaxf(ev, ALPHA * ev);                            // leaky relu
      float p = __expf(fminf(ev - rm[q], 0.f)) * bitf;       // safe, no divergence
      pA[row * 64 + kk] = f2bf(p);
    }
    // --- stage transposed Wh tile via b128 (independent temporaries) ---
    {
      int n0 = tid >> 3, ko = tid & 7;            // chunk = tid + 128q -> n = n0+16q
      size_t sbase = (size_t)n0 * (N / 8) + (k0 >> 3) + ko;
      uint4 s0 = Wt4[sbase];
      uint4 s1 = Wt4[sbase + 16 * (N / 8)];
      uint4 s2 = Wt4[sbase + 32 * (N / 8)];
      uint4 s3 = Wt4[sbase + 48 * (N / 8)];
      pB4[tid]       = s0;
      pB4[tid + 128] = s1;
      pB4[tid + 256] = s2;
      pB4[tid + 384] = s3;
    }
    __syncthreads();

    // --- two K=32 WMMAs; fragments are contiguous 16B LDS runs -> ds_load_b128 ---
#pragma unroll
    for (int kbk = 0; kbk < 2; ++kbk) {
      FragU fa, fb;
      fa.h[0] = *(const vus8*)&pA[rr * 64 + kbk * 32 + 8 * half];
      fa.h[1] = *(const vus8*)&pA[rr * 64 + kbk * 32 + 16 + 8 * half];
      int n = wv * 16 + rr;
      fb.h[0] = *(const vus8*)&pB[n * 64 + kbk * 32 + 8 * half];
      fb.h[1] = *(const vus8*)&pB[n * 64 + kbk * 32 + 16 + 8 * half];
      acc = __builtin_amdgcn_wmma_f32_16x16x32_bf16(false, fa.v, false, fb.v,
                                                    (short)0, acc, false, false);
    }
  }

  // C/D layout: VGPR v -> row (v + 8*half), col = 16*wv + rr. Normalize + elu.
#pragma unroll
  for (int v = 0; v < 8; ++v) {
    int rloc = v + 8 * half;
    float val = acc[v] * sZ[rloc];
    val = val > 0.f ? val : expm1f(val);
    out[(size_t)(i0 + rloc) * FOUT + wv * 16 + rr] = val;
  }
}

// ---------------- launch ----------------
extern "C" void kernel_launch(void* const* d_in, const int* in_sizes, int n_in,
                              void* d_out, int out_size, void* d_ws, size_t ws_size,
                              hipStream_t stream) {
  const float* h   = (const float*)d_in[0];
  const int*   adj = (const int*)  d_in[1];
  const float* M   = (const float*)d_in[2];
  const float* W   = (const float*)d_in[3];
  const float* a   = (const float*)d_in[4];
  float* out = (float*)d_out;

  // workspace carve-up (~11.2 MB)
  char* base = (char*)d_ws;
  float*          Wh    = (float*)(base);                                // 2 MB
  unsigned short* WhbfT = (unsigned short*)(base + (size_t)2*1024*1024); // 1 MB
  float* s    = (float*)(base + (size_t)3*1024*1024);                    // 5 x 32 KB
  float* u    = s + N;
  float* t    = u + N;
  float* mrow = t + N;
  float* invZ = mrow + N;
  unsigned int* bits = (unsigned int*)(invZ + N);                        // 8 MB, 16B-aligned

  k_gemm_wh  <<<(N * FOUT) / 256, 256, 0, stream>>>(h, W, Wh, WhbfT);
  k_su       <<<N / 8, 256, 0, stream>>>(Wh, a, s, u);
  k_t_bits   <<<N / 8, 256, 0, stream>>>(adj, M, u, t, (uint4*)bits);
  k_rowstats <<<N / 8, 256, 0, stream>>>((const uint4*)bits, s, t, mrow, invZ);
  k_aggregate<<<N / 16, 128, 0, stream>>>(bits, s, t, mrow, invZ, WhbfT, out);
}
